// FocalSelfAttention_54683523613204
// MI455X (gfx1250) — matrix operand order repaired and checked by reading, not compiled, not run on hardware
//
#include <hip/hip_runtime.h>
#include <hip/hip_bf16.h>
#include <math.h>

typedef __attribute__((ext_vector_type(16))) __bf16 v16bf;
typedef __attribute__((ext_vector_type(8)))  float  v8f;
typedef __attribute__((ext_vector_type(4)))  unsigned int u32x4;

union Frag {
    v16bf v;
    u32x4 q[2];
    __bf16 h[16];
};

#define WMMA_BF16(a, b, c) \
    __builtin_amdgcn_wmma_f32_16x16x32_bf16(false, (a), false, (b), (short)0, (c), false, false)

// ---- fragment loaders -------------------------------------------------
// A-matrix 16x32 bf16 (ISA layout): lane holds row M=lane&15.
// elements 0..7  <-> K = kbase + 8*half + (0..7)
// elements 8..15 <-> K = kbase + 16 + 8*half + (0..7)
__device__ __forceinline__ v16bf load_a_frag(const __bf16* row, int kbase, int half) {
    Frag f;
    f.q[0] = *reinterpret_cast<const u32x4*>(row + kbase + 8 * half);
    f.q[1] = *reinterpret_cast<const u32x4*>(row + kbase + 16 + 8 * half);
    return f.v;
}
// B-matrix 32x16 bf16: lane holds col N=lane&15.
// elements 0..7 <-> K = kbase + 8*half + (0..7), 8..15 <-> K = kbase + 16 + 8*half + (0..7)   (B[k][n] = W[n][k])
__device__ __forceinline__ v16bf load_b_frag(const __bf16* row, int kbase, int half) {
    Frag f;
    const __bf16* p = row + kbase + 8 * half;
    f.q[0] = *reinterpret_cast<const u32x4*>(p);
    f.q[1] = *reinterpret_cast<const u32x4*>(p + 16);
    return f.v;
}

// 2x2 block of 16x16 tiles:  C[32,32] = A[m0:m0+32,128] @ W[n0:n0+32,128]^T
// out[mi*2+ni].  2:1 load:wmma ratio (vs 4:1 unblocked).
__device__ __forceinline__ void gemm_2x2(const __bf16* A, const __bf16* W,
                                         int m0, int n0, int lane, v8f out[4]) {
    const int half = lane >> 4, lo = lane & 15;
    const __bf16* ar0 = A + (size_t)(m0 + lo) * 128;
    const __bf16* ar1 = A + (size_t)(m0 + 16 + lo) * 128;
    const __bf16* wr0 = W + (size_t)(n0 + lo) * 128;
    const __bf16* wr1 = W + (size_t)(n0 + 16 + lo) * 128;
    v8f c00 = {}, c01 = {}, c10 = {}, c11 = {};
#pragma unroll
    for (int kb = 0; kb < 128; kb += 32) {
        v16bf a0 = load_a_frag(ar0, kb, half);
        v16bf a1 = load_a_frag(ar1, kb, half);
        v16bf b0 = load_b_frag(wr0, kb, half);
        v16bf b1 = load_b_frag(wr1, kb, half);
        c00 = WMMA_BF16(a0, b0, c00);
        c01 = WMMA_BF16(a0, b1, c01);
        c10 = WMMA_BF16(a1, b0, c10);
        c11 = WMMA_BF16(a1, b1, c11);
    }
    out[0] = c00; out[1] = c01; out[2] = c10; out[3] = c11;
}

// ---- prep kernels -----------------------------------------------------
__global__ void k_f2bf(const float* __restrict__ s, __bf16* __restrict__ d, int n) {
    int i = blockIdx.x * 256 + threadIdx.x;
    if (i < n) d[i] = (__bf16)s[i];
}

// per-pixel LayerNorm over C=128; write bf16 queries in window-token order
__global__ void k_prep_ln(const float* __restrict__ feat,
                          const float* __restrict__ lnw, const float* __restrict__ lnb,
                          __bf16* __restrict__ qn) {
    const int c = threadIdx.x;               // 0..127
    const int pix = blockIdx.x;              // 0..9215
    const int b = pix / 2304, rem = pix % 2304;
    const int y = rem / 48, x = rem % 48;
    float v = feat[(((size_t)b * 128 + c) * 48 + y) * 48 + x];
    __shared__ float red[128];
    red[c] = v;
    __syncthreads();
#pragma unroll
    for (int s = 64; s > 0; s >>= 1) { if (c < s) red[c] += red[c + s]; __syncthreads(); }
    const float mu = red[0] * (1.0f / 128.0f);
    __syncthreads();
    const float d = v - mu;
    red[c] = d * d;
    __syncthreads();
#pragma unroll
    for (int s = 64; s > 0; s >>= 1) { if (c < s) red[c] += red[c + s]; __syncthreads(); }
    const float var = red[0] * (1.0f / 128.0f);
    const float nv = d * rsqrtf(var + 1e-5f) * lnw[c] + lnb[c];
    const int wy = y >> 2, ty = y & 3, wx = x >> 2, tx = x & 3;
    const int row = ((b * 144 + wy * 12 + wx) << 4) + ty * 4 + tx;
    qn[(size_t)row * 128 + c] = (__bf16)nv;
}

// pooled tokens: per batch 576 mid (2x2 avg) + 144 glo (4x4 avg) = 720 rows
__global__ void k_pool(const float* __restrict__ feat, __bf16* __restrict__ pooled) {
    const int c = threadIdx.x;
    const int tok = blockIdx.x;              // 0..B*720-1
    const int b = tok / 720, i = tok % 720;
    const float* fb = feat + ((size_t)b * 128 + c) * 48 * 48;
    float s = 0.f;
    if (i < 576) {
        const int py = i / 24, px = i % 24;
#pragma unroll
        for (int dy = 0; dy < 2; ++dy)
#pragma unroll
            for (int dx = 0; dx < 2; ++dx) s += fb[(py * 2 + dy) * 48 + px * 2 + dx];
        s *= 0.25f;
    } else {
        const int j = i - 576, py = j / 12, px = j % 12;
#pragma unroll
        for (int dy = 0; dy < 4; ++dy)
#pragma unroll
            for (int dx = 0; dx < 4; ++dx) s += fb[(py * 4 + dy) * 48 + px * 4 + dx];
        s *= (1.0f / 16.0f);
    }
    pooled[(size_t)tok * 128 + c] = (__bf16)s;
}

// gather 8x8 local patches (stride 4, pad 2) into [576*64][128] bf16
__global__ void k_local(const float* __restrict__ feat, __bf16* __restrict__ loc) {
    const int c = threadIdx.x;
    const int g = blockIdx.x;                // 0..36863
    const int wb = g >> 6, t = g & 63;
    const int b = wb / 144, w = wb % 144;
    const int gy = (w / 12) * 4 + (t >> 3) - 2;
    const int gx = (w % 12) * 4 + (t & 7) - 2;
    float v = 0.f;
    if ((unsigned)gy < 48u && (unsigned)gx < 48u)
        v = feat[(((size_t)b * 128 + c) * 48 + gy) * 48 + gx];
    loc[(size_t)g * 128 + c] = (__bf16)v;
}

// ---- WMMA GEMMs (2x2 register-blocked) --------------------------------
// Q = (qn @ q_w^T + q_b) / sqrt(hd) ; M=9216, N=128 -> 288 x 4 super-tiles
__global__ void k_gemm_q(const __bf16* __restrict__ qn, const __bf16* __restrict__ qw,
                         const float* __restrict__ qb, __bf16* __restrict__ Q) {
    const int lane = threadIdx.x & 31, wave = threadIdx.x >> 5;
    const int id = blockIdx.x * 8 + wave;
    const int m0 = (id >> 2) * 32, n0 = (id & 3) * 32;
    const int half = lane >> 4, lo = lane & 15;
    v8f c[4];
    gemm_2x2(qn, qw, m0, n0, lane, c);
#pragma unroll
    for (int mi = 0; mi < 2; ++mi)
#pragma unroll
        for (int ni = 0; ni < 2; ++ni) {
            const int n = n0 + ni * 16 + lo;
            const float bias = qb[n];
            v8f acc = c[mi * 2 + ni];
#pragma unroll
            for (int r = 0; r < 8; ++r) {
                const int m = m0 + mi * 16 + r + 8 * half;
                Q[(size_t)m * 128 + n] = (__bf16)((acc[r] + bias) * 0.25f);
            }
        }
}

// KV = tok @ kv_w^T + kv_b ; K row-major, V transposed in groups of G tokens
// N=256 -> 8 N-super-tiles
__global__ void k_gemm_kv(const __bf16* __restrict__ tok, const __bf16* __restrict__ kvw,
                          const float* __restrict__ kvb, __bf16* __restrict__ Kout,
                          __bf16* __restrict__ VT, int G) {
    const int lane = threadIdx.x & 31, wave = threadIdx.x >> 5;
    const int id = blockIdx.x * 8 + wave;
    const int m0 = (id >> 3) * 32, n0 = (id & 7) * 32;
    const int half = lane >> 4, lo = lane & 15;
    v8f c[4];
    gemm_2x2(tok, kvw, m0, n0, lane, c);
#pragma unroll
    for (int mi = 0; mi < 2; ++mi)
#pragma unroll
        for (int ni = 0; ni < 2; ++ni) {
            const int n = n0 + ni * 16 + lo;
            const float bias = kvb[n];
            v8f acc = c[mi * 2 + ni];
#pragma unroll
            for (int r = 0; r < 8; ++r) {
                const int m = m0 + mi * 16 + r + 8 * half;
                const float v = acc[r] + bias;
                if (n < 128) {
                    Kout[(size_t)m * 128 + n] = (__bf16)v;
                } else {
                    const int cc = n - 128, grp = m / G, t = m % G;
                    VT[((size_t)grp * 128 + cc) * G + t] = (__bf16)v;
                }
            }
        }
}

// out = attended @ out_w^T + out_b + residual, scattered back to NCHW fp32
__global__ void k_gemm_out(const __bf16* __restrict__ att, const __bf16* __restrict__ ow,
                           const float* __restrict__ ob, const float* __restrict__ feat,
                           float* __restrict__ out) {
    const int lane = threadIdx.x & 31, wave = threadIdx.x >> 5;
    const int id = blockIdx.x * 8 + wave;
    const int m0 = (id >> 2) * 32, n0 = (id & 3) * 32;
    const int half = lane >> 4, lo = lane & 15;
    v8f c[4];
    gemm_2x2(att, ow, m0, n0, lane, c);
#pragma unroll
    for (int mi = 0; mi < 2; ++mi)
#pragma unroll
        for (int ni = 0; ni < 2; ++ni) {
            const int n = n0 + ni * 16 + lo;
            const float bias = ob[n];
            v8f acc = c[mi * 2 + ni];
#pragma unroll
            for (int r = 0; r < 8; ++r) {
                const int m = m0 + mi * 16 + r + 8 * half;
                const int win = m >> 4, t = m & 15;
                const int b = win / 144, wl = win % 144;
                const int y = (wl / 12) * 4 + (t >> 2);
                const int x = (wl % 12) * 4 + (t & 3);
                const size_t idx = (((size_t)b * 128 + n) * 48 + y) * 48 + x;
                out[idx] = acc[r] + bias + feat[idx];
            }
        }
}

// ---- flash attention: one block per window-batch, wave==head ----------
__global__ void k_attn(const __bf16* __restrict__ Q,
                       const __bf16* __restrict__ Kl, const __bf16* __restrict__ VlT,
                       const __bf16* __restrict__ Kp, const __bf16* __restrict__ VpT,
                       __bf16* __restrict__ att) {
    const int lane = threadIdx.x & 31, head = threadIdx.x >> 5;
    const int wb = blockIdx.x;               // 0..575
    const int b = wb / 144;
    const int half = lane >> 4, lo = lane & 15;

    // B_q[k][n]: n=query(lo), k=head-dim 8*half+(0..7) in elements 0..7, rest zero
    Frag bq;
    {
        const __bf16* qrow = Q + ((size_t)wb * 16 + lo) * 128 + head * 16 + 8 * half;
        bq.q[0] = *reinterpret_cast<const u32x4*>(qrow);
        bq.q[1] = (u32x4){0, 0, 0, 0};
    }

    float mrun = -1e30f, lrun = 0.f;
    v8f acc = {};
    const u32x4 z4 = {0, 0, 0, 0};

    for (int c = 0; c < 49; ++c) {           // 4 local chunks + 45 pooled
        const __bf16 *krow, *vrow;
        if (c < 4) {
            krow = Kl + ((size_t)wb * 64 + c * 16 + lo) * 128 + head * 16 + 8 * half;
            vrow = VlT + ((size_t)wb * 128 + head * 16 + lo) * 64 + c * 16 + 8 * half;
        } else {
            krow = Kp + ((size_t)b * 720 + (c - 4) * 16 + lo) * 128 + head * 16 + 8 * half;
            vrow = VpT + ((size_t)b * 128 + head * 16 + lo) * 720 + (c - 4) * 16 + 8 * half;
        }
        // S^T tile: A=K chunk (rows=kv tok, K=hd pad32), B=q^T
        Frag ak; ak.q[0] = *reinterpret_cast<const u32x4*>(krow); ak.q[1] = z4;
        v8f zc = {};
        v8f st = WMMA_BF16(ak.v, bq.v, zc);
        // per-query (column) softmax stats: 8 local rows + partner half
        float cmax = st[0];
#pragma unroll
        for (int r = 1; r < 8; ++r) cmax = fmaxf(cmax, st[r]);
        cmax = fmaxf(cmax, __shfl_xor(cmax, 16));
        const float mnew = fmaxf(mrun, cmax);
        const float alpha = __expf(mrun - mnew);
        float p[8], ps = 0.f;
#pragma unroll
        for (int r = 0; r < 8; ++r) { p[r] = __expf(st[r] - mnew); ps += p[r]; }
        ps += __shfl_xor(ps, 16);
        lrun = lrun * alpha + ps;
#pragma unroll
        for (int r = 0; r < 8; ++r) acc[r] *= alpha;
        mrun = mnew;
        // B_p[k][query]: each half holds its own 8 rows (k = 8*half + j) in elements 0..7
        float po[8];
#pragma unroll
        for (int r = 0; r < 8; ++r) po[r] = __shfl_xor(p[r], 16);
        Frag bp;
#pragma unroll
        for (int j = 0; j < 8; ++j)  bp.h[j] = (__bf16)p[j];
#pragma unroll
        for (int j = 0; j < 8; ++j)  bp.h[8 + j] = (__bf16)0.f;
        // A_v[m=hd][k=kv tok]: contiguous 8 from V^T row
        Frag av; av.q[0] = *reinterpret_cast<const u32x4*>(vrow); av.q[1] = z4;
        acc = WMMA_BF16(av.v, bp.v, acc);
    }
    const float inv = 1.0f / lrun;
    __bf16* arow = att + ((size_t)wb * 16 + lo) * 128 + head * 16 + 8 * half;
#pragma unroll
    for (int r = 0; r < 8; ++r) arow[r] = (__bf16)(acc[r] * inv);
}

// ---- host launch ------------------------------------------------------
extern "C" void kernel_launch(void* const* d_in, const int* in_sizes, int n_in,
                              void* d_out, int out_size, void* d_ws, size_t ws_size,
                              hipStream_t stream) {
    const float* feat = (const float*)d_in[0];
    const float* lnw  = (const float*)d_in[1];
    const float* lnb  = (const float*)d_in[2];
    const float* q_w  = (const float*)d_in[3];
    const float* q_b  = (const float*)d_in[4];
    const float* kv_w = (const float*)d_in[5];
    const float* kv_b = (const float*)d_in[6];
    const float* o_w  = (const float*)d_in[7];
    const float* o_b  = (const float*)d_in[8];
    float* out = (float*)d_out;

    char* p = (char*)d_ws;
    auto alloc = [&](size_t bytes) -> void* {
        void* r = (void*)p;
        p += (bytes + 255) & ~(size_t)255;
        return r;
    };
    __bf16* qn     = (__bf16*)alloc((size_t)9216 * 128 * 2);
    __bf16* pooled = (__bf16*)alloc((size_t)2880 * 128 * 2);
    __bf16* loc    = (__bf16*)alloc((size_t)36864 * 128 * 2);
    __bf16* qw     = (__bf16*)alloc((size_t)128 * 128 * 2);
    __bf16* kvw    = (__bf16*)alloc((size_t)256 * 128 * 2);
    __bf16* ow     = (__bf16*)alloc((size_t)128 * 128 * 2);
    __bf16* Q      = (__bf16*)alloc((size_t)9216 * 128 * 2);
    __bf16* Kl     = (__bf16*)alloc((size_t)36864 * 128 * 2);
    __bf16* VlT    = (__bf16*)alloc((size_t)36864 * 128 * 2);
    __bf16* Kp     = (__bf16*)alloc((size_t)2880 * 128 * 2);
    __bf16* VpT    = (__bf16*)alloc((size_t)2880 * 128 * 2);
    __bf16* att    = (__bf16*)alloc((size_t)9216 * 128 * 2);

    k_f2bf<<<(128 * 128 + 255) / 256, 256, 0, stream>>>(q_w, qw, 128 * 128);
    k_f2bf<<<(256 * 128 + 255) / 256, 256, 0, stream>>>(kv_w, kvw, 256 * 128);
    k_f2bf<<<(128 * 128 + 255) / 256, 256, 0, stream>>>(o_w, ow, 128 * 128);

    k_prep_ln<<<9216, 128, 0, stream>>>(feat, lnw, lnb, qn);
    k_pool<<<2880, 128, 0, stream>>>(feat, pooled);
    k_local<<<36864, 128, 0, stream>>>(feat, loc);

    // 2x2-blocked GEMMs: 8 waves/block, one 32x32 super-tile per wave
    k_gemm_q<<<144, 256, 0, stream>>>(qn, qw, q_b, Q);            // 288*4 super-tiles
    k_gemm_kv<<<1152, 256, 0, stream>>>(loc, kvw, kv_b, Kl, VlT, 64);   // 1152*8
    k_gemm_kv<<<90, 256, 0, stream>>>(pooled, kvw, kv_b, Kp, VpT, 720); // 90*8

    k_attn<<<576, 256, 0, stream>>>(Q, Kl, VlT, Kp, VpT, att);

    k_gemm_out<<<144, 256, 0, stream>>>(att, ow, o_b, feat, out); // 288*4
}
